// MultiHeadAttentionBlock_32401233281241
// MI455X (gfx1250) — matrix operand order repaired and checked by reading, not compiled, not run on hardware
//
#include <hip/hip_runtime.h>
#include <hip/hip_bf16.h>

typedef __bf16 bf16;
typedef __attribute__((ext_vector_type(16))) __bf16 v16bf;
typedef __attribute__((ext_vector_type(8)))  float  v8f;
typedef __attribute__((ext_vector_type(4)))  int    i32x4;

#define B_  2
#define S_  2048
#define D_  1024
#define H_  16
#define DK_ 64
#define M_  (B_ * S_)   // 4096

// ---------------------------------------------------------------- helpers
static __device__ __forceinline__ v16bf frag16(uint4 lo, uint4 hi) {
  struct P { uint4 a, b; } p; p.a = lo; p.b = hi;
  return __builtin_bit_cast(v16bf, p);
}

static __device__ __forceinline__ v16bf ones_frag() {
  // B-matrix of all 1.0 bf16 (0x3F80) -> row-sum trick via WMMA
  uint4 o; o.x = o.y = o.z = o.w = 0x3F803F80u;
  struct P { uint4 a, b; } p; p.a = o; p.b = o;
  return __builtin_bit_cast(v16bf, p);
}

// CDNA5 async direct-to-LDS copy (ASYNCcnt path), with portable fallback.
#if __has_builtin(__builtin_amdgcn_global_load_async_to_lds_b128)
#define ASYNC_LDS 1
#else
#define ASYNC_LDS 0
#endif

typedef __attribute__((address_space(1))) i32x4 g_i32x4;   // global b128
typedef __attribute__((address_space(3))) i32x4 l_i32x4;   // LDS b128

static __device__ __forceinline__ void cp16(bf16* l, const bf16* g) {
#if ASYNC_LDS
  __builtin_amdgcn_global_load_async_to_lds_b128(
      (g_i32x4*)(uintptr_t)g, (l_i32x4*)(unsigned)(uintptr_t)l, 0, 0);
#else
  *(uint4*)l = *(const uint4*)g;
#endif
}

static __device__ __forceinline__ void async_wait0() {
#if __has_builtin(__builtin_amdgcn_s_wait_asynccnt)
  __builtin_amdgcn_s_wait_asynccnt(0);
#elif ASYNC_LDS
  asm volatile("s_wait_asynccnt 0" ::: "memory");
#endif
}

// 16-lane max reduction in the VALU pipe (DPP butterfly, no LDS traffic).
template <int CTRL>
static __device__ __forceinline__ float dpp_max_step(float v) {
  int o = __builtin_amdgcn_update_dpp(0, __builtin_bit_cast(int, v),
                                      CTRL, 0xF, 0xF, true);
  return fmaxf(v, __builtin_bit_cast(float, o));
}
static __device__ __forceinline__ float redmax16(float v) {
  v = dpp_max_step<0xB1>(v);    // quad_perm [1,0,3,2]  (xor 1)
  v = dpp_max_step<0x4E>(v);    // quad_perm [2,3,0,1]  (xor 2)
  v = dpp_max_step<0x141>(v);   // ROW_HALF_MIRROR      (covers 8)
  v = dpp_max_step<0x140>(v);   // ROW_MIRROR           (covers 16)
  return v;
}

// ---------------------------------------------------------------- fp32 -> bf16
__global__ __launch_bounds__(256) void cvt_f32_bf16(const float* __restrict__ src,
                                                    bf16* __restrict__ dst) {
  int i = (blockIdx.x * 256 + threadIdx.x) * 4;
  float4 f = *(const float4*)&src[i];
  struct Q { bf16 a, b, c, d; } q;
  q.a = (bf16)f.x; q.b = (bf16)f.y; q.c = (bf16)f.z; q.d = (bf16)f.w;
  *(uint2*)&dst[i] = __builtin_bit_cast(uint2, q);
}

// ---------------------------------------------------------------- bf16 GEMM
// C[m,n] = sum_k A[m,k] * W[n,k]   (A: 4096x1024, W: 1024x1024, row-major)
// MODE 0: bf16 head-split [B,H,S,DK]
// MODE 1: bf16 head-split transposed [B,H,DK,S]          (for V)
// MODE 2: f32 plain [B,S,D]                              (output projection)
// MODE 3: like MODE 0 but scaled by 1/sqrt(DK)=0.125     (for Q)
template <int MODE>
__global__ __launch_bounds__(256) void gemm_bf16(const bf16* __restrict__ A,
                                                 const bf16* __restrict__ W,
                                                 void* __restrict__ Out) {
  constexpr int LDT = 56;                  // padded row stride (halves), 16B aligned
  __shared__ __align__(16) bf16 As[2][128 * LDT];
  __shared__ __align__(16) bf16 Ws[2][128 * LDT];

  const int tid  = threadIdx.x;
  const int lane = tid & 31;
  const int wid  = tid >> 5;
  const int wm   = wid & 3;                // 4 wave-rows of 32
  const int wn   = wid >> 2;               // 2 wave-cols of 64
  const int ln   = lane & 15;
  const int hi   = lane >> 4;

  const int n0 = blockIdx.x * 128;
  const int m0 = blockIdx.y * 128;

  auto stage = [&](int kk, int buf) {      // 2 x 16B per thread per tile
    #pragma unroll
    for (int c = 0; c < 2; ++c) {
      int idx  = tid + c * 256;            // 0..511
      int row  = idx >> 2;
      int part = (idx & 3) * 8;            // halves
      cp16(&As[buf][row * LDT + part], &A[(m0 + row) * D_ + kk + part]);
      cp16(&Ws[buf][row * LDT + part], &W[(n0 + row) * D_ + kk + part]);
    }
  };

  v8f acc[2][4];
  #pragma unroll
  for (int x = 0; x < 2; ++x)
    #pragma unroll
    for (int y = 0; y < 4; ++y) acc[x][y] = {};

  stage(0, 0);
  constexpr int NK = D_ / 32;
  for (int j = 0; j < NK; ++j) {
    async_wait0();
    __syncthreads();                       // tile j visible; prior reads done
    if (j + 1 < NK) stage((j + 1) * 32, (j + 1) & 1);

    const bf16* Ab = As[j & 1];
    const bf16* Wb = Ws[j & 1];
    v16bf af[2], wf[4];
    #pragma unroll
    for (int mi = 0; mi < 2; ++mi) {
      int row = wm * 32 + mi * 16 + ln;
      af[mi] = frag16(*(const uint4*)&Ab[row * LDT + 8 * hi],
                      *(const uint4*)&Ab[row * LDT + 16 + 8 * hi]);
    }
    #pragma unroll
    for (int ni = 0; ni < 4; ++ni) {
      int row = wn * 64 + ni * 16 + ln;
      wf[ni] = frag16(*(const uint4*)&Wb[row * LDT + 8 * hi],
                      *(const uint4*)&Wb[row * LDT + 16 + 8 * hi]);
    }
    #pragma unroll
    for (int mi = 0; mi < 2; ++mi)
      #pragma unroll
      for (int ni = 0; ni < 4; ++ni)
        acc[mi][ni] = __builtin_amdgcn_wmma_f32_16x16x32_bf16(
            false, af[mi], false, wf[ni], (short)0, acc[mi][ni], false, false);
  }

  #pragma unroll
  for (int mi = 0; mi < 2; ++mi)
    #pragma unroll
    for (int ni = 0; ni < 4; ++ni)
      #pragma unroll
      for (int i = 0; i < 8; ++i) {
        int m = m0 + wm * 32 + mi * 16 + 8 * hi + i;
        int n = n0 + wn * 64 + ni * 16 + ln;
        float v = acc[mi][ni][i];
        if (MODE == 3) v *= 0.125f;        // fold softmax 1/sqrt(DK) into Q
        if (MODE == 2) {
          ((float*)Out)[m * D_ + n] = v;
        } else {
          int bb = m >> 11, s = m & (S_ - 1);
          int h  = n >> 6,  dk = n & 63;
          bf16* o = (bf16*)Out;
          if (MODE == 1)
            o[(((bb * H_ + h) * DK_) + dk) * S_ + s] = (bf16)v;
          else
            o[(((bb * H_ + h) * S_) + s) * DK_ + dk] = (bf16)v;
        }
      }
}

// ---------------------------------------------------------------- flash attention
// One workgroup = (b, h, 64-query block); 4 waves x 16 query rows each.
__global__ __launch_bounds__(128) void attn_fused(const bf16* __restrict__ HQ,
                                                  const bf16* __restrict__ HK,
                                                  const bf16* __restrict__ HVT,
                                                  bf16* __restrict__ AV) {
  constexpr int LDK = 72;                  // 32 x 64 K-tile, padded row stride
  constexpr int LDV = 56;                  // 64 x 32 V^T-tile, padded row stride
  __shared__ __align__(16) bf16 Ks[2][32 * LDK];
  __shared__ __align__(16) bf16 Vs[2][64 * LDV];
  __shared__ __align__(16) bf16 Ps[4][16 * 32];   // per-wave P transpose scratch

  const int tid  = threadIdx.x;
  const int lane = tid & 31;
  const int wid  = tid >> 5;
  const int ln   = lane & 15;
  const int hi   = lane >> 4;

  const int q0 = blockIdx.x * 64;
  const int h  = blockIdx.y;
  const int b  = blockIdx.z;

  const int qk_base = ((b * H_ + h) * S_) * DK_;
  const int vt_base = ((b * H_ + h) * DK_) * S_;

  auto stage = [&](int k0, int buf) {
    #pragma unroll
    for (int c = 0; c < 2; ++c) {
      int idx = tid + c * 128;
      { int row = idx >> 3, part = (idx & 7) * 8;   // K tile: 32 x 64 halves
        cp16(&Ks[buf][row * LDK + part],
             &HK[qk_base + (k0 + row) * DK_ + part]); }
      { int row = idx >> 2, part = (idx & 3) * 8;   // V^T tile: 64 x 32 halves
        cp16(&Vs[buf][row * LDV + part],
             &HVT[vt_base + row * S_ + k0 + part]); }
    }
  };

  // Q A-fragments (16 rows x 64 d, pre-scaled by 0.125 at projection time)
  const int wrow0 = q0 + wid * 16;         // wave's min query row (uniform/wave)
  const int qrow  = wrow0 + ln;
  v16bf qf[2];
  #pragma unroll
  for (int s = 0; s < 2; ++s) {
    const bf16* p = HQ + qk_base + qrow * DK_ + 32 * s;
    qf[s] = frag16(*(const uint4*)&p[8 * hi], *(const uint4*)&p[16 + 8 * hi]);
  }
  const v16bf onesf = ones_frag();
  const int rowbase = wrow0 + 8 * hi;      // this lane's first of 8 rows

  v8f O[4];
  #pragma unroll
  for (int c = 0; c < 4; ++c) O[c] = {};
  float mrow[8], lrow[8];
  #pragma unroll
  for (int i = 0; i < 8; ++i) { mrow[i] = -1e30f; lrow[i] = 0.f; }

  const int nblk = (q0 >> 5) + 2;          // causal: keys up to q0+63
  stage(0, 0);
  for (int j = 0; j < nblk; ++j) {
    const int k0 = j * 32;
    async_wait0();
    __syncthreads();                       // tile j ready; prior reads done
    if (j + 1 < nblk) stage((j + 1) * 32, (j + 1) & 1);

    if (k0 > wrow0 + 15) continue;         // fully-masked for this wave (uniform)
    const bool need_mask = (k0 + 31 > wrow0);   // diagonal block? (uniform/wave)
    const bf16* Kb = Ks[j & 1];
    const bf16* Vb = Vs[j & 1];

    // scores: 16q x 32k, two 16x16 WMMA tiles, K-dim 64 = 2 slices
    float st[2][8];
    #pragma unroll
    for (int t = 0; t < 2; ++t) {
      v16bf bk[2];
      #pragma unroll
      for (int s = 0; s < 2; ++s) {
        int row = 16 * t + ln;
        bk[s] = frag16(*(const uint4*)&Kb[row * LDK + 32 * s + 8 * hi],
                       *(const uint4*)&Kb[row * LDK + 32 * s + 16 + 8 * hi]);
      }
      v8f sf = {};
      sf = __builtin_amdgcn_wmma_f32_16x16x32_bf16(false, qf[0], false, bk[0],
                                                   (short)0, sf, false, false);
      sf = __builtin_amdgcn_wmma_f32_16x16x32_bf16(false, qf[1], false, bk[1],
                                                   (short)0, sf, false, false);
      if (need_mask) {
        int key = k0 + 16 * t + ln;
        #pragma unroll
        for (int i = 0; i < 8; ++i)
          st[t][i] = (key > rowbase + i) ? -1e9f : sf[i];
      } else {
        #pragma unroll
        for (int i = 0; i < 8; ++i) st[t][i] = sf[i];
      }
    }

    // online softmax; row max via DPP butterfly (VALU only, no LDS)
    #pragma unroll
    for (int i = 0; i < 8; ++i) {
      float mv    = redmax16(fmaxf(st[0][i], st[1][i]));
      float mnew  = fmaxf(mrow[i], mv);
      float alpha = __expf(mrow[i] - mnew);
      float p0 = __expf(st[0][i] - mnew);
      float p1 = __expf(st[1][i] - mnew);
      lrow[i] *= alpha;
      mrow[i]  = mnew;
      #pragma unroll
      for (int c = 0; c < 4; ++c) O[c][i] *= alpha;
      int prow = 8 * hi + i;               // transpose P into A-frag layout via LDS
      Ps[wid][prow * 32 + ln]      = (bf16)p0;
      Ps[wid][prow * 32 + 16 + ln] = (bf16)p1;
    }
    __builtin_amdgcn_wave_barrier();       // wave-private LDS: order stores/loads
    __asm__ volatile("" ::: "memory");

    v16bf pf = frag16(*(const uint4*)&Ps[wid][ln * 32 + 8 * hi],
                      *(const uint4*)&Ps[wid][ln * 32 + 16 + 8 * hi]);

    // row sums via WMMA against all-ones B (C-frag layout matches lrow[i])
    v8f Ls = {};
    Ls = __builtin_amdgcn_wmma_f32_16x16x32_bf16(false, pf, false, onesf,
                                                 (short)0, Ls, false, false);
    #pragma unroll
    for (int i = 0; i < 8; ++i) lrow[i] += Ls[i];

    #pragma unroll
    for (int c = 0; c < 4; ++c) {          // O(16x64) += P(16x32) @ V(32x64)
      int row = 16 * c + ln;
      v16bf bv = frag16(*(const uint4*)&Vb[row * LDV + 8 * hi],
                        *(const uint4*)&Vb[row * LDV + 16 + 8 * hi]);
      O[c] = __builtin_amdgcn_wmma_f32_16x16x32_bf16(false, pf, false, bv,
                                                     (short)0, O[c], false, false);
    }
  }

  #pragma unroll
  for (int c = 0; c < 4; ++c)
    #pragma unroll
    for (int i = 0; i < 8; ++i) {
      int row = rowbase + i;
      int col = h * DK_ + 16 * c + ln;
      AV[(b * S_ + row) * D_ + col] = (bf16)(O[c][i] / lrow[i]);
    }
}

// ---------------------------------------------------------------- launch
extern "C" void kernel_launch(void* const* d_in, const int* in_sizes, int n_in,
                              void* d_out, int out_size, void* d_ws, size_t ws_size,
                              hipStream_t stream) {
  (void)in_sizes; (void)n_in; (void)out_size; (void)ws_size;
  const float* q  = (const float*)d_in[0];
  const float* k  = (const float*)d_in[1];
  const float* v  = (const float*)d_in[2];
  // d_in[3] = causal mask (applied analytically)
  const float* wq = (const float*)d_in[4];
  const float* wk = (const float*)d_in[5];
  const float* wv = (const float*)d_in[6];
  const float* wo = (const float*)d_in[7];

  char* ws = (char*)d_ws;
  const size_t MB = 1u << 20;
  bf16* qb  = (bf16*)(ws +  0 * MB);   // 8 MB each
  bf16* kb  = (bf16*)(ws +  8 * MB);
  bf16* vb  = (bf16*)(ws + 16 * MB);
  bf16* wqb = (bf16*)(ws + 24 * MB);   // 2 MB each
  bf16* wkb = (bf16*)(ws + 26 * MB);
  bf16* wvb = (bf16*)(ws + 28 * MB);
  bf16* wob = (bf16*)(ws + 30 * MB);
  bf16* HQ  = (bf16*)(ws + 32 * MB);   // [B,H,S,DK] bf16 (Q pre-scaled), 8 MB
  bf16* HK  = (bf16*)(ws + 40 * MB);   // [B,H,S,DK]
  bf16* HVT = (bf16*)(ws + 48 * MB);   // [B,H,DK,S]
  bf16* AVb = (bf16*)(ws + 56 * MB);   // [B,S,D]

  cvt_f32_bf16<<<M_ * D_ / 1024, 256, 0, stream>>>(q, qb);
  cvt_f32_bf16<<<M_ * D_ / 1024, 256, 0, stream>>>(k, kb);
  cvt_f32_bf16<<<M_ * D_ / 1024, 256, 0, stream>>>(v, vb);
  cvt_f32_bf16<<<D_ * D_ / 1024, 256, 0, stream>>>(wq, wqb);
  cvt_f32_bf16<<<D_ * D_ / 1024, 256, 0, stream>>>(wk, wkb);
  cvt_f32_bf16<<<D_ * D_ / 1024, 256, 0, stream>>>(wv, wvb);
  cvt_f32_bf16<<<D_ * D_ / 1024, 256, 0, stream>>>(wo, wob);

  dim3 ggrid(D_ / 128, M_ / 128);                       // 8 x 32
  gemm_bf16<3><<<ggrid, 256, 0, stream>>>(qb, wqb, (void*)HQ);   // Q (scaled)
  gemm_bf16<0><<<ggrid, 256, 0, stream>>>(kb, wkb, (void*)HK);   // K
  gemm_bf16<1><<<ggrid, 256, 0, stream>>>(vb, wvb, (void*)HVT);  // V^T

  attn_fused<<<dim3(S_ / 64, H_, B_), 128, 0, stream>>>(HQ, HK, HVT, AVb);

  gemm_bf16<2><<<ggrid, 256, 0, stream>>>(AVb, wob, d_out);      // out proj
}
